// StandardTRM_7954279432809
// MI455X (gfx1250) — matrix operand order repaired and checked by reading, not compiled, not run on hardware
//
#include <hip/hip_runtime.h>

#define HID 256
#define FF 1024
#define BATCH_TOTAL 131072
#define ROWS_PER_WG 128

typedef __attribute__((ext_vector_type(16))) __bf16 v16bf;
typedef __attribute__((ext_vector_type(8)))  float  v8f;
typedef __attribute__((ext_vector_type(4)))  int    i32x4;

// ---------------------------------------------------------------------------
// CDNA5 async global->LDS copy (no data VGPRs, tracked by ASYNCcnt).
// Builtin signature (from compiler diagnostic): b128 variant takes
// (int4 addrspace(1)*, int4 addrspace(3)*, imm offset, imm cpol).
// ---------------------------------------------------------------------------
#if __has_builtin(__builtin_amdgcn_global_load_async_to_lds_b128) && \
    __has_builtin(__builtin_amdgcn_s_wait_asynccnt)
#define HAVE_ASYNC_LDS 1
#else
#define HAVE_ASYNC_LDS 0
#endif

typedef __attribute__((address_space(1))) i32x4* as1_i32x4_p;
typedef __attribute__((address_space(3))) i32x4* as3_i32x4_p;

static __device__ __forceinline__ void async_copy16(const __bf16* g, __bf16* l) {
#if HAVE_ASYNC_LDS
  __builtin_amdgcn_global_load_async_to_lds_b128((as1_i32x4_p)g, (as3_i32x4_p)l, 0, 0);
#else
  *(uint4*)l = *(const uint4*)g;
#endif
}

static __device__ __forceinline__ void async_wait_all() {
#if HAVE_ASYNC_LDS
  __builtin_amdgcn_s_wait_asynccnt(0);
#endif
}

// ---------------------------------------------------------------------------
// WMMA helper: D = A(16x32 bf16) * B(32x16 bf16) + C(16x16 f32)
// ---------------------------------------------------------------------------
static __device__ __forceinline__ v8f wmma_bf16(v16bf a, v16bf b, v8f c) {
  return __builtin_amdgcn_wmma_f32_16x16x32_bf16(false, a, false, b, (short)0, c,
                                                 false, false);
}

// A-fragment (16x32, 16-bit) from a row-major bf16 tile:
//   lanes 0-15 : row m=l16, halfs 0..15 = K kbase+0..15
//   lanes 16-31: row m=l16, halfs 0..15 = K kbase+16..31
static __device__ __forceinline__ v16bf afrag(const __bf16* base, int m, int kbase,
                                              int stride, int hi8) {
  const __bf16* p = base + m * stride + kbase + 2 * hi8;
  union { v16bf v; uint4 q[2]; } u;
  u.q[0] = *(const uint4*)(p);
  u.q[1] = *(const uint4*)(p + 8);
  return u.v;
}

// B-fragment (32x16, 16-bit): per-lane 16 contiguous K values.
// p must already point at  base + n*stride + kbase + (hiHalf ? 16 : 0)
static __device__ __forceinline__ v16bf bfrag(const __bf16* p) {
  union { v16bf v; uint4 q[2]; } u;
  u.q[0] = *(const uint4*)(p);
  u.q[1] = *(const uint4*)(p + 8);
  return u.v;
}

// A-fragment built from global f32 rows (x matrix), converting to bf16.
static __device__ __forceinline__ v16bf afrag_f32g(const float* p) {
  v16bf r;
#pragma unroll
  for (int i = 0; i < 8; ++i) r[i] = (__bf16)p[i];
#pragma unroll
  for (int i = 0; i < 8; ++i) r[i + 8] = (__bf16)p[i + 8];
  return r;
}

static __device__ __forceinline__ float gelu_exact(float x) {
  return 0.5f * x * (1.0f + erff(x * 0.70710678118654752f));
}

// reduce across lanes {0..15} and {16..31} independently (xor masks < 16)
static __device__ __forceinline__ float hsum16(float v) {
  v += __shfl_xor(v, 1, 32);
  v += __shfl_xor(v, 2, 32);
  v += __shfl_xor(v, 4, 32);
  v += __shfl_xor(v, 8, 32);
  return v;
}

// ---------------------------------------------------------------------------
// Prologue: transpose f32 [K][N] -> bf16 [Npad][K]   (rows >= N zero padded)
// ---------------------------------------------------------------------------
__global__ void transpose_cvt(const float* __restrict__ src, __bf16* __restrict__ dst,
                              int K, int N, int Npad) {
  int idx = blockIdx.x * blockDim.x + threadIdx.x;
  if (idx >= K * Npad) return;
  int n = idx / K;
  int k = idx - n * K;
  float v = (n < N) ? src[k * N + n] : 0.0f;
  dst[n * K + k] = (__bf16)v;
}

// ---------------------------------------------------------------------------
// Fused TRM kernel: 128 rows per workgroup, 16 rows per wave32.
// LDS (dynamic, 256 KB): z[128][256] bf16 | y[128][256] bf16 |
//                        stage 64KB (double buffered weights) | h[8][16][256] bf16
// ---------------------------------------------------------------------------
__global__ __launch_bounds__(256, 1) void trm_fused(
    const float* __restrict__ x,
    const float* __restrict__ b_in, const float* __restrict__ y_init,
    const float* __restrict__ b_up, const float* __restrict__ b_down,
    const float* __restrict__ ln_w, const float* __restrict__ ln_b,
    const float* __restrict__ b_comb, const float* __restrict__ b_yt,
    const float* __restrict__ b_head,
    const __bf16* __restrict__ WinT, const __bf16* __restrict__ WupT,
    const __bf16* __restrict__ WdownT, const __bf16* __restrict__ WcombT,
    const __bf16* __restrict__ WytT, const __bf16* __restrict__ WheadT,
    float* __restrict__ out) {
  extern __shared__ __bf16 smem[];
  __bf16* zL  = smem;              // 32768 halfs
  __bf16* yL  = smem + 32768;      // 32768 halfs
  __bf16* stg = smem + 65536;      // 32768 halfs (staging, double buffered)
  __bf16* hSc = smem + 98304;      // 8 waves * 16*256 halfs

  const int tid  = threadIdx.x;
  const int wave = tid >> 5;
  const int lane = tid & 31;
  const int l16  = lane & 15;
  const int hi8  = (lane >> 4) << 3;  // 0 or 8
  const int hi16 = hi8 << 1;          // 0 or 16
  const int wRow = wave << 4;
  const int rowBase = blockIdx.x * ROWS_PER_WG;

  __bf16* zW = zL + wRow * HID;
  __bf16* yW = yL + wRow * HID;
  __bf16* hW = hSc + wave * (16 * HID);

  __bf16* stgU0 = stg;           __bf16* stgU1 = stg + 8192;
  __bf16* stgD0 = stg + 16384;   __bf16* stgD1 = stg + 24576;
  __bf16* stgC0 = stg;           __bf16* stgC1 = stg + 16384;

  // ------------------ y = broadcast(y_init) ------------------
  {
    __bf16 bv = (__bf16)y_init[tid];  // tid == column (blockDim == HID)
    for (int r = 0; r < ROWS_PER_WG; ++r) yL[r * HID + tid] = bv;
  }

  // ------------------ z = x @ W_in + b_in ------------------
  {
    v16bf xA[8];
    const float* xrow = x + (size_t)(rowBase + wRow + l16) * HID;
#pragma unroll
    for (int ks = 0; ks < 8; ++ks) xA[ks] = afrag_f32g(xrow + ks * 32 + hi16);

    {  // stage chunk 0: 64 rows of WinT (64*256 halfs)
      const __bf16* s = WinT + (tid >> 2) * HID + (tid & 3) * 64;
      __bf16* d = stgC0 + tid * 64;
#pragma unroll
      for (int i = 0; i < 8; ++i) async_copy16(s + i * 8, d + i * 8);
    }
    async_wait_all();
    __syncthreads();
    for (int c = 0; c < 4; ++c) {
      if (c + 1 < 4) {  // async prefetch next chunk into the free buffer
        const __bf16* s = WinT + ((c + 1) * 64 + (tid >> 2)) * HID + (tid & 3) * 64;
        __bf16* d = ((c & 1) ? stgC0 : stgC1) + tid * 64;
#pragma unroll
        for (int i = 0; i < 8; ++i) async_copy16(s + i * 8, d + i * 8);
      }
      const __bf16* sc = (c & 1) ? stgC1 : stgC0;
#pragma unroll
      for (int t = 0; t < 4; ++t) {
        v8f a = {};
#pragma unroll
        for (int ks = 0; ks < 8; ++ks) {
          v16bf b = bfrag(sc + (t * 16 + l16) * HID + ks * 32 + hi16);
          a = wmma_bf16(xA[ks], b, a);
        }
        const int n0 = c * 64 + t * 16;
        const float bi = b_in[n0 + l16];
#pragma unroll
        for (int r = 0; r < 8; ++r)
          zW[(r + hi8) * HID + n0 + l16] = (__bf16)(a[r] + bi);
      }
      async_wait_all();
      __syncthreads();
    }
  }

  // ------------------ recurrent H / L cycles ------------------
  for (int hc = 0; hc < 3; ++hc) {
    for (int lc = 0; lc < 6; ++lc) {
      {  // stage chunk 0 (up rows 0..31, down k 0..31)
        const __bf16* us = WupT + (tid >> 3) * HID + (tid & 7) * 32;
        __bf16* ud = stgU0 + tid * 32;
        async_copy16(us, ud);
        async_copy16(us + 8, ud + 8);
        const __bf16* ds = WdownT + tid * FF;
        __bf16* dd = stgD0 + tid * 32;
#pragma unroll
        for (int i = 0; i < 4; ++i) async_copy16(ds + i * 8, dd + i * 8);
      }
      v8f acc[16];
#pragma unroll
      for (int t = 0; t < 16; ++t) acc[t] = v8f{};
      async_wait_all();
      __syncthreads();

      for (int c = 0; c < 32; ++c) {   // 32 chunks of 32 FFN columns
        if (c + 1 < 32) {  // async prefetch next up/down chunk
          const __bf16* us =
              WupT + ((c + 1) * 32 + (tid >> 3)) * HID + (tid & 7) * 32;
          __bf16* ud = ((c & 1) ? stgU0 : stgU1) + tid * 32;
          async_copy16(us, ud);
          async_copy16(us + 8, ud + 8);
          const __bf16* ds = WdownT + tid * FF + (c + 1) * 32;
          __bf16* dd = ((c & 1) ? stgD0 : stgD1) + tid * 32;
#pragma unroll
          for (int i = 0; i < 4; ++i) async_copy16(ds + i * 8, dd + i * 8);
        }
        const __bf16* su = (c & 1) ? stgU1 : stgU0;
        const __bf16* sd = (c & 1) ? stgD1 : stgD0;

        v8f h0 = {}, h1 = {};
#pragma unroll
        for (int ks = 0; ks < 8; ++ks) {
          v16bf a  = afrag(zW, l16, ks * 32, HID, hi8);
          v16bf b0 = bfrag(su + l16 * HID + ks * 32 + hi16);
          v16bf b1 = bfrag(su + (16 + l16) * HID + ks * 32 + hi16);
          h0 = wmma_bf16(a, b0, h0);
          h1 = wmma_bf16(a, b1, h1);
        }
        const float bu0 = b_up[c * 32 + l16];
        const float bu1 = b_up[c * 32 + 16 + l16];
#pragma unroll
        for (int r = 0; r < 8; ++r) {  // C-layout -> LDS bounce (wave-private)
          const int m = r + hi8;
          hW[m * 32 + l16]      = (__bf16)gelu_exact(h0[r] + bu0);
          hW[m * 32 + 16 + l16] = (__bf16)gelu_exact(h1[r] + bu1);
        }
        v16bf hA = afrag(hW, l16, 0, 32, hi8);  // same-wave DS ops are in order
#pragma unroll
        for (int t = 0; t < 16; ++t) {
          v16bf bd = bfrag(sd + (t * 16 + l16) * 32 + hi16);
          acc[t] = wmma_bf16(hA, bd, acc[t]);
        }
        async_wait_all();
        __syncthreads();
      }

      // residual + layernorm, fully in registers + half-wave shuffles
#pragma unroll
      for (int r = 0; r < 8; ++r) {
        const int m = r + hi8;
        float v[16];
        float s = 0.f;
#pragma unroll
        for (int t = 0; t < 16; ++t) {
          v[t] = acc[t][r] + b_down[t * 16 + l16] + (float)zW[m * HID + t * 16 + l16];
          s += v[t];
        }
        s = hsum16(s);
        const float mean = s * (1.0f / 256.0f);
        float q = 0.f;
#pragma unroll
        for (int t = 0; t < 16; ++t) { const float d = v[t] - mean; q += d * d; }
        q = hsum16(q);
        const float rstd = rsqrtf(q * (1.0f / 256.0f) + 1e-5f);
#pragma unroll
        for (int t = 0; t < 16; ++t)
          zW[m * HID + t * 16 + l16] =
              (__bf16)((v[t] - mean) * rstd * ln_w[t * 16 + l16] + ln_b[t * 16 + l16]);
      }
    }  // lc

    // ------------- h = gelu([z,y] @ W_comb + b_comb) -------------
    {
      const __bf16* s = WcombT + (tid >> 3) * 512 + (tid & 7) * 64;
      __bf16* d = stgC0 + tid * 64;
#pragma unroll
      for (int i = 0; i < 8; ++i) async_copy16(s + i * 8, d + i * 8);
    }
    async_wait_all();
    __syncthreads();
    for (int c = 0; c < 8; ++c) {  // 8 chunks of 32 output columns, K = 512
      if (c + 1 < 8) {
        const __bf16* s =
            WcombT + ((c + 1) * 32 + (tid >> 3)) * 512 + (tid & 7) * 64;
        __bf16* d = ((c & 1) ? stgC0 : stgC1) + tid * 64;
#pragma unroll
        for (int i = 0; i < 8; ++i) async_copy16(s + i * 8, d + i * 8);
      }
      const __bf16* scb = (c & 1) ? stgC1 : stgC0;
      v8f h0 = {}, h1 = {};
#pragma unroll
      for (int ks = 0; ks < 16; ++ks) {
        v16bf a = (ks < 8) ? afrag(zW, l16, ks * 32, HID, hi8)
                           : afrag(yW, l16, (ks - 8) * 32, HID, hi8);
        v16bf b0 = bfrag(scb + l16 * 512 + ks * 32 + hi16);
        v16bf b1 = bfrag(scb + (16 + l16) * 512 + ks * 32 + hi16);
        h0 = wmma_bf16(a, b0, h0);
        h1 = wmma_bf16(a, b1, h1);
      }
      const float bc0 = b_comb[c * 32 + l16];
      const float bc1 = b_comb[c * 32 + 16 + l16];
#pragma unroll
      for (int r = 0; r < 8; ++r) {
        const int m = r + hi8;
        hW[m * HID + c * 32 + l16]      = (__bf16)gelu_exact(h0[r] + bc0);
        hW[m * HID + c * 32 + 16 + l16] = (__bf16)gelu_exact(h1[r] + bc1);
      }
      async_wait_all();
      __syncthreads();
    }

    // ------------- y += h @ W_yt + b_yt -------------
    {
      const __bf16* s = WytT + tid * HID;
      __bf16* d = stgU0 + tid * 32;
#pragma unroll
      for (int i = 0; i < 4; ++i) async_copy16(s + i * 8, d + i * 8);
    }
    v8f acc2[16];
#pragma unroll
    for (int t = 0; t < 16; ++t) acc2[t] = v8f{};
    async_wait_all();
    __syncthreads();
    for (int kc = 0; kc < 8; ++kc) {
      if (kc + 1 < 8) {
        const __bf16* s = WytT + tid * HID + (kc + 1) * 32;
        __bf16* d = ((kc & 1) ? stgU0 : stgU1) + tid * 32;
#pragma unroll
        for (int i = 0; i < 4; ++i) async_copy16(s + i * 8, d + i * 8);
      }
      const __bf16* sy = (kc & 1) ? stgU1 : stgU0;
      v16bf hA = afrag(hW, l16, kc * 32, HID, hi8);
#pragma unroll
      for (int t = 0; t < 16; ++t) {
        v16bf bb = bfrag(sy + (t * 16 + l16) * 32 + hi16);
        acc2[t] = wmma_bf16(hA, bb, acc2[t]);
      }
      async_wait_all();
      __syncthreads();
    }
#pragma unroll
    for (int t = 0; t < 16; ++t) {
      const float by = b_yt[t * 16 + l16];
#pragma unroll
      for (int r = 0; r < 8; ++r) {
        const int m = r + hi8;
        const float v = acc2[t][r] + by + (float)yW[m * HID + t * 16 + l16];
        yW[m * HID + t * 16 + l16] = (__bf16)v;
      }
    }
  }  // hc

  // ------------------ out = y @ W_head + b_head ------------------
  {
    v8f o = {};
#pragma unroll
    for (int ks = 0; ks < 8; ++ks) {
      v16bf a = afrag(yW, l16, ks * 32, HID, hi8);
      v16bf b = bfrag(WheadT + l16 * HID + ks * 32 + hi16);  // 16x256, rows>=10 zero
      o = wmma_bf16(a, b, o);
    }
    if (l16 < 10) {
      const float bh = b_head[l16];
#pragma unroll
      for (int r = 0; r < 8; ++r) {
        const int m = r + hi8;
        out[(size_t)(rowBase + wRow + m) * 10 + l16] = o[r] + bh;
      }
    }
  }
}

// ---------------------------------------------------------------------------
extern "C" void kernel_launch(void* const* d_in, const int* in_sizes, int n_in,
                              void* d_out, int out_size, void* d_ws, size_t ws_size,
                              hipStream_t stream) {
  (void)in_sizes; (void)n_in; (void)out_size; (void)ws_size;
  const float* x      = (const float*)d_in[0];
  const float* W_in   = (const float*)d_in[1];
  const float* b_in   = (const float*)d_in[2];
  const float* y_init = (const float*)d_in[3];
  const float* W_up   = (const float*)d_in[4];
  const float* b_up   = (const float*)d_in[5];
  const float* W_down = (const float*)d_in[6];
  const float* b_down = (const float*)d_in[7];
  const float* ln_w   = (const float*)d_in[8];
  const float* ln_b   = (const float*)d_in[9];
  const float* W_comb = (const float*)d_in[10];
  const float* b_comb = (const float*)d_in[11];
  const float* W_yt   = (const float*)d_in[12];
  const float* b_yt   = (const float*)d_in[13];
  const float* W_head = (const float*)d_in[14];
  const float* b_head = (const float*)d_in[15];
  float* out = (float*)d_out;

  __bf16* ws = (__bf16*)d_ws;
  __bf16* WinT   = ws;                          // [256][256]
  __bf16* WupT   = ws + 65536;                  // [1024][256]
  __bf16* WdownT = WupT + 262144;               // [256][1024]
  __bf16* WcombT = WdownT + 262144;             // [256][512]
  __bf16* WytT   = WcombT + 131072;             // [256][256]
  __bf16* WheadT = WytT + 65536;                // [16][256] (zero padded)

  auto tl = [&](const float* src, __bf16* dst, int K, int N, int Npad) {
    int total = K * Npad;
    transpose_cvt<<<(total + 255) / 256, 256, 0, stream>>>(src, dst, K, N, Npad);
  };
  tl(W_in,   WinT,   256,  256,  256);
  tl(W_up,   WupT,   256,  1024, 1024);
  tl(W_down, WdownT, 1024, 256,  256);
  tl(W_comb, WcombT, 512,  256,  256);
  tl(W_yt,   WytT,   256,  256,  256);
  tl(W_head, WheadT, 256,  10,   16);

  const size_t lds_bytes = (size_t)131072 * sizeof(__bf16);  // 256 KB dynamic LDS
  trm_fused<<<BATCH_TOTAL / ROWS_PER_WG, 256, lds_bytes, stream>>>(
      x, b_in, y_init, b_up, b_down, ln_w, ln_b, b_comb, b_yt, b_head,
      WinT, WupT, WdownT, WcombT, WytT, WheadT, out);
}
